// ModifiedAttention_25194278159047
// MI455X (gfx1250) — matrix operand order repaired and checked by reading, not compiled, not run on hardware
//
#include <hip/hip_runtime.h>
#include <hip/hip_bf16.h>
#include <math.h>

typedef __attribute__((ext_vector_type(16))) _Float16 v16h;
typedef __attribute__((ext_vector_type(8)))  float    v8f;

constexpr int NB  = 4;
constexpr int NS  = 1024;
constexpr int ND  = 768;
constexpr int NH  = 24;
constexpr int NDH = 64;
constexpr int NBS = NB * NS;          // 4096
constexpr int NHD = NH * NDH;         // 1536

// ---------------------------------------------------------------------------
// Capability tiers for tile fill:
//   GA_TDM==1 : ROCm 7.2 lane, 5-arg tensor_load_to_lds
//   GA_TDM==2 : amdgpu-toolchain lane (TDM header present), 6-arg builtin
//   GA_ASYNC  : async global->LDS per-lane copies (ASYNCcnt)
//   else      : plain vector copy through VGPRs
// ---------------------------------------------------------------------------
#if __has_builtin(__builtin_amdgcn_tensor_load_to_lds) && \
    __has_builtin(__builtin_amdgcn_s_wait_tensorcnt)
#  if __has_include(<hip/amd_detail/amd_gfx1250_TDM.h>)
#    define GA_TDM 2
#  else
#    define GA_TDM 1
#  endif
#else
#  define GA_TDM 0
#endif

#if !GA_TDM && __has_builtin(__builtin_amdgcn_global_load_async_to_lds_b128) && \
    __has_builtin(__builtin_amdgcn_s_wait_asynccnt)
#  define GA_ASYNC 1
#else
#  define GA_ASYNC 0
#endif

#if GA_TDM
typedef __attribute__((ext_vector_type(4))) unsigned int v4u_t;
typedef __attribute__((ext_vector_type(8))) int          v8i_t;
typedef __attribute__((ext_vector_type(4))) int          v4i_t;

// 2D f16 tile DMA: global (row stride gstride elems) -> LDS rows padded by
// (pad_amt+1) dwords after every 2^(pad_int+1) dwords (reproduces +8-half pad).
static __device__ inline void ga_tdm_load(const _Float16* gsrc, void* ldst,
                                          int tile_w, int rows, int gstride,
                                          int pad_int, int pad_amt) {
  const unsigned long long ga = (unsigned long long)(size_t)gsrc;
  const unsigned lds =
      (unsigned)(unsigned long long)(__attribute__((address_space(3))) char*)ldst;
  v4u_t g0;
  g0[0] = 1u;                                            // count=1 (valid D#)
  g0[1] = lds;                                           // lds_addr
  g0[2] = (unsigned)ga;                                  // global_addr[31:0]
  g0[3] = ((unsigned)(ga >> 32) & 0x01ffffffu) | 0x80000000u; // addr[56:32]|type=2
  v8i_t g1;
  g1[0] = (int)((1u << 16) |                             // data_size = 2B
                (1u << 20) |                             // pad_enable
                ((unsigned)pad_int << 22) |              // pad_interval
                ((unsigned)pad_amt << 25));              // pad_amount
  g1[1] = (int)((unsigned)tile_w << 16);                 // tensor_dim0[15:0]
  g1[2] = (int)((unsigned)rows << 16);                   // tensor_dim1[15:0]
  g1[3] = (int)((unsigned)tile_w << 16);                 // tile_dim0
  g1[4] = (int)(unsigned)rows;                           // tile_dim1 (tile_dim2=0)
  g1[5] = (int)gstride;                                  // tensor_dim0_stride
  g1[6] = 0;
  g1[7] = 0;
  v4i_t gz;
  gz[0] = gz[1] = gz[2] = gz[3] = 0;
#if GA_TDM == 1
  __builtin_amdgcn_tensor_load_to_lds(g0, g1, gz, gz, 0);
#else
  v8i_t gz8;
#pragma unroll
  for (int i = 0; i < 8; ++i) gz8[i] = 0;
  __builtin_amdgcn_tensor_load_to_lds(g0, g1, gz, gz, gz8, 0);
#endif
}
#endif  // GA_TDM

#if GA_ASYNC
static __device__ inline void ga_async_cp16(void* lds, const void* g) {
  __builtin_amdgcn_global_load_async_to_lds_b128(
      (__attribute__((address_space(1))) void*)const_cast<void*>(g),
      (__attribute__((address_space(3))) void*)lds, 0, 0);
}
#endif

// ---------------------------------------------------------------------------
// WMMA helpers (CDNA5 gfx1250, wave32)
// ---------------------------------------------------------------------------
static __device__ inline v8f v8f_zero() {
  v8f v;
#pragma unroll
  for (int i = 0; i < 8; ++i) v[i] = 0.0f;
  return v;
}

static __device__ inline v8f wmma_f16(v16h a, v16h b, v8f c) {
  return __builtin_amdgcn_wmma_f32_16x16x32_f16(
      false, a, false, b, (short)0, c, false, false);
}

struct alignas(16) ga_frag_bits { uint4 a, b; };

// A fragment 16x32 (MxK) from row-major LDS tile; ldm multiple of 8.
static __device__ inline v16h load_a_frag(const _Float16* base, int ldm) {
  const int lane = threadIdx.x & 31;
  const int half = lane >> 4;
  const int m    = lane & 15;
  const _Float16* p = base + m * ldm + half * 8;
  ga_frag_bits u;
  u.a = *(const uint4*)p;
  u.b = *(const uint4*)(p + 16);
  return __builtin_bit_cast(v16h, u);
}

// B fragment 32x16 (KxN), element (k,n) at base[n*ldn + k] (n-major storage).
static __device__ inline v16h load_b_fragT(const _Float16* base, int ldn) {
  const int lane = threadIdx.x & 31;
  const int half = lane >> 4;
  const int n    = lane & 15;
  const _Float16* p = base + n * ldn + half * 8;
  ga_frag_bits u;
  u.a = *(const uint4*)p;
  u.b = *(const uint4*)(p + 16);
  return __builtin_bit_cast(v16h, u);
}

static __device__ inline _Float16 h_lo(unsigned u) {
  return __builtin_bit_cast(_Float16, (unsigned short)(u & 0xffffu));
}
static __device__ inline _Float16 h_hi(unsigned u) {
  return __builtin_bit_cast(_Float16, (unsigned short)(u >> 16));
}

// ---------------------------------------------------------------------------
// Prep kernels
// ---------------------------------------------------------------------------
__global__ void ga_conv_x(const float* __restrict__ x, _Float16* __restrict__ xh, int n) {
  for (int i = blockIdx.x * blockDim.x + threadIdx.x; i < n; i += gridDim.x * blockDim.x)
    xh[i] = (_Float16)x[i];
}

__global__ void ga_norms(const float* __restrict__ WV, const float* __restrict__ WO,
                         float* __restrict__ nv, float* __restrict__ no) {
  const int t = blockIdx.x * blockDim.x + threadIdx.x;
  if (t >= NH * NDH) return;
  const int h = t >> 6, dh = t & 63;
  float s = 0.0f;
  const float* p = WV + (size_t)h * ND * NDH + dh;
  for (int d = 0; d < ND; ++d) { const float w = p[(size_t)d * NDH]; s += w * w; }
  nv[t] = fmaxf(sqrtf(s), 1e-12f);
  s = 0.0f;
  const float* p2 = WO + ((size_t)h * NDH + dh) * ND;
  for (int m = 0; m < ND; ++m) { const float w = p2[m]; s += w * w; }
  no[t] = fmaxf(sqrtf(s), 1e-12f);
}

__global__ void ga_conv_w(const float* __restrict__ WQ, const float* __restrict__ WK,
                          const float* __restrict__ WV, const float* __restrict__ WO,
                          const float* __restrict__ nv, const float* __restrict__ no,
                          _Float16* __restrict__ wq, _Float16* __restrict__ wk,
                          _Float16* __restrict__ wv, _Float16* __restrict__ wo) {
  const int n = NH * ND * NDH;
  for (int i = blockIdx.x * blockDim.x + threadIdx.x; i < n; i += gridDim.x * blockDim.x) {
    wq[i] = (_Float16)WQ[i];
    wk[i] = (_Float16)WK[i];
    const int dh = i & 63;
    const int h  = (i >> 6) / ND;
    wv[i] = (_Float16)(WV[i] / nv[h * NDH + dh]);   // [h][d][dh]
    wo[i] = (_Float16)(WO[i] / no[i / ND]);         // [h][dh][m]
  }
}

__global__ void __launch_bounds__(256) ga_gates(
    const float* __restrict__ x,
    const float* __restrict__ WgQ, const float* __restrict__ WgK,
    const float* __restrict__ bgQ, const float* __restrict__ bgK,
    float* __restrict__ qg, float* __restrict__ kg) {
  const int bs = blockIdx.x;
  const int b = bs >> 10, s = bs & 1023;
  __shared__ float xr[ND];
  for (int i = threadIdx.x; i < ND; i += 256) xr[i] = x[(size_t)bs * ND + i];
  __syncthreads();
  const int t = threadIdx.x;
  if (t < 2 * NH) {
    const int h = t % NH, which = t / NH;
    const float* w = (which == 0 ? WgQ : WgK) + (size_t)h * ND;
    float acc = 0.0f;
    for (int d = 0; d < ND; ++d) acc += xr[d] * w[d];
    acc += (which == 0 ? bgQ[h] : bgK[h]);
    float* dst = (which == 0 ? qg : kg);
    dst[((size_t)b * NH + h) * NS + s] = acc;
  }
}

// ---------------------------------------------------------------------------
// QKV projection GEMM: grid (BS/64, H, 3), 128 threads = 4 waves.
// ---------------------------------------------------------------------------
__global__ void __launch_bounds__(128) ga_proj(
    const _Float16* __restrict__ xh,
    const _Float16* __restrict__ wq, const _Float16* __restrict__ wk,
    const _Float16* __restrict__ wv,
    const float* __restrict__ bq, const float* __restrict__ bk,
    const float* __restrict__ bv,
    _Float16* __restrict__ qo, _Float16* __restrict__ ko, _Float16* __restrict__ vo) {
  const int mblk = blockIdx.x;
  const int h    = blockIdx.y;
  const int proj = blockIdx.z;
  const _Float16* w    = (proj == 0) ? wq : (proj == 1) ? wk : wv;
  const float*    bias = (proj == 0) ? bq : (proj == 1) ? bk : bv;
  _Float16*       outp = (proj == 0) ? qo : (proj == 1) ? ko : vo;

  __shared__ _Float16 xt[64][40];    // [row][k]  (natural, +8 pad)
  __shared__ _Float16 wtT[64][40];   // [n][k]    (transposed staging)

  const int tid = threadIdx.x;
  const int wave = tid >> 5;
  const int lane = tid & 31, half = lane >> 4, nlo = lane & 15;

  v8f acc[4];
#pragma unroll
  for (int i = 0; i < 4; ++i) acc[i] = v8f_zero();

  for (int k0 = 0; k0 < ND; k0 += 32) {
#if GA_TDM
    if (tid < 32)
      ga_tdm_load(xh + (size_t)mblk * 64 * ND + k0, &xt[0][0], 32, 64, ND, 3, 3);
#else
    for (int i = tid; i < 256; i += 128) {
      const int r = i >> 2, c = (i & 3) << 3;
      const _Float16* g = &xh[((size_t)mblk * 64 + r) * ND + k0 + c];
#if GA_ASYNC
      ga_async_cp16(&xt[r][c], g);
#else
      *(uint4*)&xt[r][c] = *(const uint4*)g;
#endif
    }
#endif
    // W tile transposed into LDS: (k,n) -> wtT[n][k]
    for (int i = tid; i < 512; i += 128) {
      const int kk = i >> 4, n4 = (i & 15) << 2;
      const uint2 d = *(const uint2*)&w[((size_t)h * ND + k0 + kk) * NDH + n4];
      wtT[n4 + 0][kk] = h_lo(d.x);
      wtT[n4 + 1][kk] = h_hi(d.x);
      wtT[n4 + 2][kk] = h_lo(d.y);
      wtT[n4 + 3][kk] = h_hi(d.y);
    }
#if GA_TDM
    if (tid < 32) __builtin_amdgcn_s_wait_tensorcnt(0);
#elif GA_ASYNC
    __builtin_amdgcn_s_wait_asynccnt(0);
#endif
    __syncthreads();
    const v16h a = load_a_frag(&xt[wave * 16][0], 40);
#pragma unroll
    for (int nf = 0; nf < 4; ++nf) {
      const v16h bf = load_b_fragT(&wtT[nf * 16][0], 40);
      acc[nf] = wmma_f16(a, bf, acc[nf]);
    }
    __syncthreads();
  }

#pragma unroll
  for (int nf = 0; nf < 4; ++nf) {
    const float bval = bias[h * NDH + nf * 16 + nlo];
#pragma unroll
    for (int r = 0; r < 8; ++r) {
      const int row = mblk * 64 + wave * 16 + r + 8 * half;   // bs index
      const int b = row >> 10, s = row & 1023;
      outp[(((size_t)b * NH + h) * NS + s) * NDH + nf * 16 + nlo] =
          (_Float16)(acc[nf][r] + bval);
    }
  }
}

// ---------------------------------------------------------------------------
// Gated flash attention: grid (S/64, H, B), 128 threads = 4 waves.
// ---------------------------------------------------------------------------
__global__ void __launch_bounds__(128) ga_attn(
    const _Float16* __restrict__ qh, const _Float16* __restrict__ kh,
    const _Float16* __restrict__ vh,
    const float* __restrict__ qg, const float* __restrict__ kg,
    _Float16* __restrict__ zh) {
  const int qt = blockIdx.x;
  const int h  = blockIdx.y;
  const int b  = blockIdx.z;
  const size_t bh = (size_t)b * NH + h;
  const _Float16* qbase = qh + bh * NS * NDH;
  const _Float16* kbase = kh + bh * NS * NDH;
  const _Float16* vbase = vh + bh * NS * NDH;
  const float* qgb = qg + bh * NS;
  const float* kgb = kg + bh * NS;

  __shared__ _Float16 qtl[64][72];          // [q][dh]   natural
  __shared__ _Float16 ktl[64][72];          // [key][dh] natural (B^T source)
  __shared__ _Float16 vtlT[64][72];         // [dh][key] transposed staging
  __shared__ float qgv[64];
  __shared__ float kgv[64];
  __shared__ _Float16 pscr[4][16][72];      // per-wave P relayout scratch

  const int tid = threadIdx.x;
  const int wave = tid >> 5;
  const int lane = tid & 31, half = lane >> 4, nlo = lane & 15;

#if GA_TDM
  if (tid < 32)
    ga_tdm_load(qbase + (size_t)qt * 64 * NDH, &qtl[0][0], 64, 64, 64, 4, 3);
#else
  for (int i = tid; i < 512; i += 128) {
    const int r = i >> 3, c = (i & 7) << 3;
    const _Float16* g = &qbase[((size_t)qt * 64 + r) * NDH + c];
#if GA_ASYNC
    ga_async_cp16(&qtl[r][c], g);
#else
    *(uint4*)&qtl[r][c] = *(const uint4*)g;
#endif
  }
#endif
  if (tid < 64) qgv[tid] = qgb[qt * 64 + tid];

  float m_acc[8], l_acc[8];
#pragma unroll
  for (int r = 0; r < 8; ++r) { m_acc[r] = -1e30f; l_acc[r] = 0.0f; }
  v8f o[4];
#pragma unroll
  for (int i = 0; i < 4; ++i) o[i] = v8f_zero();

  for (int kt = 0; kt <= qt; ++kt) {
    __syncthreads();   // previous tiles fully consumed
#if GA_TDM
    if (tid < 32)
      ga_tdm_load(kbase + (size_t)kt * 64 * NDH, &ktl[0][0], 64, 64, 64, 4, 3);
#else
    for (int i = tid; i < 512; i += 128) {
      const int r = i >> 3, c = (i & 7) << 3;
      const _Float16* g = &kbase[((size_t)kt * 64 + r) * NDH + c];
#if GA_ASYNC
      ga_async_cp16(&ktl[r][c], g);
#else
      *(uint4*)&ktl[r][c] = *(const uint4*)g;
#endif
    }
#endif
    // V tile transposed into LDS: (key,dh) -> vtlT[dh][key]
    for (int i = tid; i < 512; i += 128) {
      const int key = i >> 4, d4 = (i & 15) << 2;
      const uint2 dv = *(const uint2*)&vbase[((size_t)kt * 64 + key) * NDH + d4];
      vtlT[d4 + 0][key] = h_lo(dv.x);
      vtlT[d4 + 1][key] = h_hi(dv.x);
      vtlT[d4 + 2][key] = h_lo(dv.y);
      vtlT[d4 + 3][key] = h_hi(dv.y);
    }
    if (tid < 64) kgv[tid] = kgb[kt * 64 + tid];
#if GA_TDM
    if (tid < 32) __builtin_amdgcn_s_wait_tensorcnt(0);
#elif GA_ASYNC
    __builtin_amdgcn_s_wait_asynccnt(0);
#endif
    __syncthreads();

    // ---- S = Q * K^T ----
    v8f sf[4];
#pragma unroll
    for (int i = 0; i < 4; ++i) sf[i] = v8f_zero();
#pragma unroll
    for (int kc = 0; kc < 2; ++kc) {
      const v16h a = load_a_frag(&qtl[wave * 16][kc * 32], 72);
#pragma unroll
      for (int nf = 0; nf < 4; ++nf) {
        const v16h bf = load_b_fragT(&ktl[nf * 16][kc * 32], 72);
        sf[nf] = wmma_f16(a, bf, sf[nf]);
      }
    }

    // ---- scale + causal mask ----
    const bool diag = (kt == qt);
#pragma unroll
    for (int nf = 0; nf < 4; ++nf) {
      const int kpos = kt * 64 + nf * 16 + nlo;
#pragma unroll
      for (int r = 0; r < 8; ++r) {
        const int qpos = qt * 64 + wave * 16 + r + 8 * half;
        float s = sf[nf][r] * 0.125f;
        if (diag && kpos > qpos) s = -1e30f;
        sf[nf][r] = s;
      }
    }

    // ---- online softmax ----
    float rm[8];
#pragma unroll
    for (int r = 0; r < 8; ++r)
      rm[r] = fmaxf(fmaxf(sf[0][r], sf[1][r]), fmaxf(sf[2][r], sf[3][r]));
#pragma unroll
    for (int m = 8; m >= 1; m >>= 1)
#pragma unroll
      for (int r = 0; r < 8; ++r) rm[r] = fmaxf(rm[r], __shfl_xor(rm[r], m, 32));

    float alpha[8];
#pragma unroll
    for (int r = 0; r < 8; ++r) {
      const float mnew = fmaxf(m_acc[r], rm[r]);
      alpha[r] = __expf(m_acc[r] - mnew);
      m_acc[r] = mnew;
    }

    float rs[8];
#pragma unroll
    for (int r = 0; r < 8; ++r) rs[r] = 0.0f;
#pragma unroll
    for (int nf = 0; nf < 4; ++nf)
#pragma unroll
      for (int r = 0; r < 8; ++r) {
        const float p = __expf(sf[nf][r] - m_acc[r]);
        sf[nf][r] = p;
        rs[r] += p;
      }
#pragma unroll
    for (int m = 8; m >= 1; m >>= 1)
#pragma unroll
      for (int r = 0; r < 8; ++r) rs[r] += __shfl_xor(rs[r], m, 32);
#pragma unroll
    for (int r = 0; r < 8; ++r) l_acc[r] = l_acc[r] * alpha[r] + rs[r];

    // ---- gate*P -> per-wave LDS scratch (A relayout) ----
#pragma unroll
    for (int nf = 0; nf < 4; ++nf) {
      const float kgn = kgv[nf * 16 + nlo];
#pragma unroll
      for (int r = 0; r < 8; ++r) {
        float g = qgv[wave * 16 + r + 8 * half] * kgn;
        g = fminf(fmaxf(g, 0.0f), 1.0f);
        pscr[wave][r + 8 * half][nf * 16 + nlo] = (_Float16)(sf[nf][r] * g);
      }
    }

    // ---- O = O*alpha + (gate*P) @ V ----
#pragma unroll
    for (int nf = 0; nf < 4; ++nf)
#pragma unroll
      for (int r = 0; r < 8; ++r) o[nf][r] *= alpha[r];
#pragma unroll
    for (int kc = 0; kc < 2; ++kc) {
      const v16h a = load_a_frag(&pscr[wave][0][kc * 32], 72);
#pragma unroll
      for (int nf = 0; nf < 4; ++nf) {
        const v16h bf = load_b_fragT(&vtlT[nf * 16][kc * 32], 72);
        o[nf] = wmma_f16(a, bf, o[nf]);
      }
    }
  }

  // ---- z = O / l ----
#pragma unroll
  for (int r = 0; r < 8; ++r) l_acc[r] = 1.0f / l_acc[r];
#pragma unroll
  for (int nf = 0; nf < 4; ++nf)
#pragma unroll
    for (int r = 0; r < 8; ++r) {
      const int row = qt * 64 + wave * 16 + r + 8 * half;
      zh[((size_t)b * NS + row) * NHD + h * NDH + nf * 16 + nlo] =
          (_Float16)(o[nf][r] * l_acc[r]);
    }
}

// ---------------------------------------------------------------------------
// Output projection: out[BSx768] = z[BSx1536] @ Wocat[1536x768]
// ---------------------------------------------------------------------------
__global__ void __launch_bounds__(128) ga_outproj(
    const _Float16* __restrict__ zh, const _Float16* __restrict__ woh,
    float* __restrict__ out) {
  const int mblk = blockIdx.x;
  const int nblk = blockIdx.y;

  __shared__ _Float16 zt[64][40];    // [row][k]
  __shared__ _Float16 wtT[64][40];   // [n][k]

  const int tid = threadIdx.x;
  const int wave = tid >> 5;
  const int lane = tid & 31, half = lane >> 4, nlo = lane & 15;

  v8f acc[4];
#pragma unroll
  for (int i = 0; i < 4; ++i) acc[i] = v8f_zero();

  for (int k0 = 0; k0 < NHD; k0 += 32) {
#if GA_TDM
    if (tid < 32)
      ga_tdm_load(zh + (size_t)mblk * 64 * NHD + k0, &zt[0][0], 32, 64, NHD, 3, 3);
#else
    for (int i = tid; i < 256; i += 128) {
      const int r = i >> 2, c = (i & 3) << 3;
      const _Float16* g = &zh[((size_t)mblk * 64 + r) * NHD + k0 + c];
#if GA_ASYNC
      ga_async_cp16(&zt[r][c], g);
#else
      *(uint4*)&zt[r][c] = *(const uint4*)g;
#endif
    }
#endif
    for (int i = tid; i < 512; i += 128) {
      const int kk = i >> 4, n4 = (i & 15) << 2;
      const uint2 d = *(const uint2*)&woh[((size_t)k0 + kk) * ND + nblk * 64 + n4];
      wtT[n4 + 0][kk] = h_lo(d.x);
      wtT[n4 + 1][kk] = h_hi(d.x);
      wtT[n4 + 2][kk] = h_lo(d.y);
      wtT[n4 + 3][kk] = h_hi(d.y);
    }
#if GA_TDM
    if (tid < 32) __builtin_amdgcn_s_wait_tensorcnt(0);
#elif GA_ASYNC
    __builtin_amdgcn_s_wait_asynccnt(0);
#endif
    __syncthreads();
    const v16h a = load_a_frag(&zt[wave * 16][0], 40);
#pragma unroll
    for (int nf = 0; nf < 4; ++nf) {
      const v16h bf = load_b_fragT(&wtT[nf * 16][0], 40);
      acc[nf] = wmma_f16(a, bf, acc[nf]);
    }
    __syncthreads();
  }

#pragma unroll
  for (int nf = 0; nf < 4; ++nf)
#pragma unroll
    for (int r = 0; r < 8; ++r) {
      const int row = mblk * 64 + wave * 16 + r + 8 * half;
      out[(size_t)row * ND + nblk * 64 + nf * 16 + nlo] = acc[nf][r];
    }
}

// ---------------------------------------------------------------------------
// Host-side launch
// ---------------------------------------------------------------------------
extern "C" void kernel_launch(void* const* d_in, const int* in_sizes, int n_in,
                              void* d_out, int out_size, void* d_ws, size_t ws_size,
                              hipStream_t stream) {
  (void)in_sizes; (void)n_in; (void)out_size; (void)ws_size;

  const float* x   = (const float*)d_in[0];
  const float* WQ  = (const float*)d_in[1];
  const float* WK  = (const float*)d_in[2];
  const float* WV  = (const float*)d_in[3];
  const float* WO  = (const float*)d_in[4];
  const float* bQ  = (const float*)d_in[5];
  const float* bK  = (const float*)d_in[6];
  const float* bV  = (const float*)d_in[7];
  const float* WgQ = (const float*)d_in[8];
  const float* WgK = (const float*)d_in[9];
  const float* bgQ = (const float*)d_in[10];
  const float* bgK = (const float*)d_in[11];
  float* out = (float*)d_out;

  char* ws = (char*)d_ws;
  size_t off = 0;
  auto carve = [&](size_t bytes) -> char* {
    char* p = ws + off;
    off = (off + bytes + 255) & ~(size_t)255;
    return p;
  };

  _Float16* xh  = (_Float16*)carve((size_t)NBS * ND * 2);
  _Float16* wqh = (_Float16*)carve((size_t)NH * ND * NDH * 2);
  _Float16* wkh = (_Float16*)carve((size_t)NH * ND * NDH * 2);
  _Float16* wvh = (_Float16*)carve((size_t)NH * ND * NDH * 2);
  _Float16* woh = (_Float16*)carve((size_t)NH * NDH * ND * 2);
  _Float16* qh  = (_Float16*)carve((size_t)NB * NH * NS * NDH * 2);
  _Float16* kh  = (_Float16*)carve((size_t)NB * NH * NS * NDH * 2);
  _Float16* vhh = (_Float16*)carve((size_t)NB * NH * NS * NDH * 2);
  float*    qgp = (float*)carve((size_t)NB * NH * NS * 4);
  float*    kgp = (float*)carve((size_t)NB * NH * NS * 4);
  _Float16* zhp = (_Float16*)carve((size_t)NBS * NHD * 2);
  float*    nv  = (float*)carve((size_t)NH * NDH * 4);
  float*    no  = (float*)carve((size_t)NH * NDH * 4);

  ga_conv_x<<<3072, 256, 0, stream>>>(x, xh, NBS * ND);
  ga_norms<<<6, 256, 0, stream>>>(WV, WO, nv, no);
  ga_conv_w<<<1152, 256, 0, stream>>>(WQ, WK, WV, WO, nv, no, wqh, wkh, wvh, woh);
  ga_gates<<<NBS, 256, 0, stream>>>(x, WgQ, WgK, bgQ, bgK, qgp, kgp);
  ga_proj<<<dim3(NBS / 64, NH, 3), 128, 0, stream>>>(
      xh, wqh, wkh, wvh, bQ, bK, bV, qh, kh, vhh);
  ga_attn<<<dim3(NS / 64, NH, NB), 128, 0, stream>>>(qh, kh, vhh, qgp, kgp, zhp);
  ga_outproj<<<dim3(NBS / 64, ND / 64), 128, 0, stream>>>(zhp, woh, out);
}
